// Quantizer_10565619548577
// MI455X (gfx1250) — matrix operand order repaired and checked by reading, not compiled, not run on hardware
//
#include <hip/hip_runtime.h>
#include <hip/hip_bf16.h>
#include <math.h>

// ---------- sizes ----------
#define N_ROWS   65536      // 32*2048 flattened z rows
#define E_DIM    256
#define N_CODES  1024
#define BETA     0.25f
#define LAMBDA   0.99f

// output layout (floats, concatenated in reference return order)
#define O_LOSS   0
#define O_ZQ     1
#define O_IDX    (1 + (size_t)N_ROWS * E_DIM)                 // 16777217
#define O_PERP   (O_IDX + N_ROWS)                             // 16842753
#define O_NT     (O_PERP + 1)                                 // 16842754
#define O_MT     (O_NT + N_CODES)                             // 16843778

typedef __attribute__((ext_vector_type(16))) __bf16 v16bf;
typedef __attribute__((ext_vector_type(8)))  __bf16 v8bf;
typedef __attribute__((ext_vector_type(8)))  float  v8f;

__device__ __forceinline__ unsigned short f2bf(float f) {
    unsigned u = __float_as_uint(f);
    unsigned r = u + 0x7FFFu + ((u >> 16) & 1u);   // round-to-nearest-even
    return (unsigned short)(r >> 16);
}
__device__ __forceinline__ float bf2f(unsigned short h) {
    return __uint_as_float(((unsigned)h) << 16);
}

// ---------- K0: embedding -> split bf16 + exact f32 row norms ----------
__global__ __launch_bounds__(256)
void prep_emb_kernel(const float* __restrict__ emb,
                     unsigned short* __restrict__ ehi,
                     unsigned short* __restrict__ elo,
                     float* __restrict__ e_norm) {
    const int j = blockIdx.x, t = threadIdx.x;
    const int i = j * E_DIM + t;
    float v = emb[i];
    unsigned short h = f2bf(v);
    ehi[i] = h;
    elo[i] = f2bf(v - bf2f(h));
    __shared__ float red[256];
    red[t] = v * v;
    __syncthreads();
    for (int s = 128; s > 0; s >>= 1) {
        if (t < s) red[t] += red[t + s];
        __syncthreads();
    }
    if (t == 0) e_norm[j] = red[0];
}

// ---------- K1: zero accumulators ----------
__global__ void zero_kernel(float* __restrict__ p, int n) {
    int i = blockIdx.x * 256 + threadIdx.x;
    if (i < n) p[i] = 0.0f;
}

// ---------- K2: fused distance GEMM (bf16x3 WMMA) + argmin ----------
// 128 threads (4 waves), 64 z-rows per workgroup, all 1024 codes swept.
__global__ __launch_bounds__(128)
void argmin_kernel(const float* __restrict__ z,
                   const unsigned short* __restrict__ ehi,
                   const unsigned short* __restrict__ elo,
                   const float* __restrict__ e_norm,
                   int*   __restrict__ idx_ws,
                   float* __restrict__ idx_outf) {
    __shared__ unsigned short sHi[64 * E_DIM];   // 32 KB
    __shared__ unsigned short sLo[64 * E_DIM];   // 32 KB

    const int tid = threadIdx.x;
    const size_t rowBase = (size_t)blockIdx.x * 64;
    const float* zb = z + rowBase * E_DIM;

    // Stage: 64 rows of f32 z -> split bf16 in LDS (z read once from HBM)
    for (int i = tid; i < 64 * E_DIM; i += 128) {
        float v = zb[i];
        unsigned short h = f2bf(v);
        sHi[i] = h;
        sLo[i] = f2bf(v - bf2f(h));
    }
    __syncthreads();

    const int wave = tid >> 5;
    const int lane = tid & 31;
    const int half = lane >> 4;    // K-half selector (A) / K-group (B)
    const int ln   = lane & 15;    // A row M / B column N

    // A-matrix lane row (ISA 16-bit A 16x32 layout: M = lane%16)
    const unsigned short* aHiRow = &sHi[(wave * 16 + ln) * E_DIM];
    const unsigned short* aLoRow = &sLo[(wave * 16 + ln) * E_DIM];

    float minv[8];
    int   mini[8];
#pragma unroll
    for (int r = 0; r < 8; ++r) { minv[r] = 3.402823466e38f; mini[r] = 0; }

    for (int nt = 0; nt < 64; ++nt) {
        const int code = nt * 16 + ln;
        const unsigned short* bh = ehi + code * E_DIM + half * 8;
        const unsigned short* bl = elo + code * E_DIM + half * 8;
        v8f acc = {0.f, 0.f, 0.f, 0.f, 0.f, 0.f, 0.f, 0.f};

#pragma unroll
        for (int ks = 0; ks < 8; ++ks) {
            const int kb = ks * 32;
            // A: two 8-element contiguous K chunks per lane (ISA striping)
            v8bf h0 = *(const v8bf*)(aHiRow + kb + half * 8);
            v8bf h1 = *(const v8bf*)(aHiRow + kb + 16 + half * 8);
            v8bf l0 = *(const v8bf*)(aLoRow + kb + half * 8);
            v8bf l1 = *(const v8bf*)(aLoRow + kb + 16 + half * 8);
            v16bf aH = __builtin_shufflevector(h0, h1, 0,1,2,3,4,5,6,7,8,9,10,11,12,13,14,15);
            v16bf aL = __builtin_shufflevector(l0, l1, 0,1,2,3,4,5,6,7,8,9,10,11,12,13,14,15);
            // B: two 8-element K chunks for column `ln` (same striping as A)
            v16bf bH = __builtin_shufflevector(*(const v8bf*)(bh + kb), *(const v8bf*)(bh + kb + 16), 0,1,2,3,4,5,6,7,8,9,10,11,12,13,14,15);
            v16bf bL = __builtin_shufflevector(*(const v8bf*)(bl + kb), *(const v8bf*)(bl + kb + 16), 0,1,2,3,4,5,6,7,8,9,10,11,12,13,14,15);
            // bf16x3 error-compensated accumulation: hi*hi + hi*lo + lo*hi
            acc = __builtin_amdgcn_wmma_f32_16x16x32_bf16(false, aH, false, bH, (short)0, acc, false, false);
            acc = __builtin_amdgcn_wmma_f32_16x16x32_bf16(false, aH, false, bL, (short)0, acc, false, false);
            acc = __builtin_amdgcn_wmma_f32_16x16x32_bf16(false, aL, false, bH, (short)0, acc, false, false);
        }

        const float en = e_norm[code];
#pragma unroll
        for (int r = 0; r < 8; ++r) {
            float d = en - 2.0f * acc[r];        // ||z||^2 dropped: row-constant
            if (d < minv[r]) { minv[r] = d; mini[r] = code; }  // '<' keeps first index
        }
    }

    // Cross-lane min+argmin over the 16 columns (lanes sharing half,r share row M)
#pragma unroll
    for (int r = 0; r < 8; ++r) {
        float v = minv[r];
        int   i = mini[r];
#pragma unroll
        for (int m = 1; m < 16; m <<= 1) {
            float ov = __shfl_xor(v, m, 32);
            int   oi = __shfl_xor(i, m, 32);
            if (ov < v || (ov == v && oi < i)) { v = ov; i = oi; }
        }
        if (ln == 0) {
            size_t row = rowBase + wave * 16 + half * 8 + r;   // C layout: M = half*8 + r
            idx_ws[row]   = i;
            idx_outf[row] = (float)i;
        }
    }
}

// ---------- K3: gather z_q, straight-through output, loss + segment sums ----------
__global__ __launch_bounds__(256)
void gather_kernel(const float* __restrict__ z,
                   const float* __restrict__ emb,
                   const int* __restrict__ idx_ws,
                   float* __restrict__ zq_out,
                   float* __restrict__ sums,
                   float* __restrict__ counts,
                   float* __restrict__ loss_acc) {
    const int row = blockIdx.x, t = threadIdx.x;
    const int c = idx_ws[row];
    const size_t i = (size_t)row * E_DIM + t;
    float zv = z[i];
    float e  = emb[c * E_DIM + t];
    zq_out[i] = zv + (e - zv);                 // straight-through, same rounding as ref
    float diff = e - zv;
    atomicAdd(&sums[c * E_DIM + t], zv);       // segment_sum(zf)
    __shared__ float red[256];
    red[t] = diff * diff;
    __syncthreads();
    for (int s = 128; s > 0; s >>= 1) {
        if (t < s) red[t] += red[t + s];
        __syncthreads();
    }
    if (t == 0) {
        atomicAdd(loss_acc, red[0]);
        atomicAdd(&counts[c], 1.0f);           // segment_sum(ones)
    }
}

// ---------- K4a: N_t EMA + perplexity + loss finalize (1 block, 1024 threads) ----------
__global__ __launch_bounds__(1024)
void finalize_nt_kernel(const float* __restrict__ N_t,
                        const float* __restrict__ counts,
                        const float* __restrict__ loss_acc,
                        float* __restrict__ out) {
    const int j = threadIdx.x;
    float c = counts[j];
    out[O_NT + j] = (c > 0.0f) ? (N_t[j] * LAMBDA + c * (1.0f - LAMBDA)) : N_t[j];
    float p = c * (1.0f / (float)N_ROWS);
    __shared__ float red[1024];
    red[j] = p * logf(p + 1e-10f);
    __syncthreads();
    for (int s = 512; s > 0; s >>= 1) {
        if (j < s) red[j] += red[j + s];
        __syncthreads();
    }
    if (j == 0) {
        out[O_PERP] = expf(-red[0]);
        out[O_LOSS] = BETA * loss_acc[0] * (1.0f / ((float)N_ROWS * (float)E_DIM));
    }
}

// ---------- K4b: m_t EMA ----------
__global__ __launch_bounds__(256)
void finalize_mt_kernel(const float* __restrict__ m_t,
                        const float* __restrict__ sums,
                        const float* __restrict__ counts,
                        float* __restrict__ out_mt) {
    const int j = blockIdx.x, t = threadIdx.x;
    const size_t i = (size_t)j * E_DIM + t;
    float sel = counts[j];
    out_mt[i] = (sel > 0.0f) ? (m_t[i] * LAMBDA + sums[i] * (1.0f - LAMBDA)) : m_t[i];
}

extern "C" void kernel_launch(void* const* d_in, const int* in_sizes, int n_in,
                              void* d_out, int out_size, void* d_ws, size_t ws_size,
                              hipStream_t stream) {
    const float* z    = (const float*)d_in[0];
    const float* emb  = (const float*)d_in[1];
    const float* N_t  = (const float*)d_in[2];
    const float* m_t  = (const float*)d_in[3];
    float* out = (float*)d_out;

    // workspace carve-up (~2.3 MB)
    char* w = (char*)d_ws;
    int*            idx_ws = (int*)w;            w += (size_t)N_ROWS * sizeof(int);
    float*          e_norm = (float*)w;          w += (size_t)N_CODES * sizeof(float);
    unsigned short* ehi    = (unsigned short*)w; w += (size_t)N_CODES * E_DIM * 2;
    unsigned short* elo    = (unsigned short*)w; w += (size_t)N_CODES * E_DIM * 2;
    float*          counts = (float*)w;          w += (size_t)N_CODES * sizeof(float);
    float*          sums   = (float*)w;          w += (size_t)N_CODES * E_DIM * sizeof(float);
    float*          loss_acc = (float*)w;        w += sizeof(float);

    // K0: embedding split + norms
    prep_emb_kernel<<<N_CODES, 256, 0, stream>>>(emb, ehi, elo, e_norm);
    // K1: zero counts+sums+loss (contiguous)
    const int nz = N_CODES + N_CODES * E_DIM + 1;
    zero_kernel<<<(nz + 255) / 256, 256, 0, stream>>>(counts, nz);
    // K2: WMMA distance + argmin
    argmin_kernel<<<N_ROWS / 64, 128, 0, stream>>>(z, ehi, elo, e_norm,
                                                   idx_ws, out + O_IDX);
    // K3: gather, straight-through, loss, segment sums
    gather_kernel<<<N_ROWS, 256, 0, stream>>>(z, emb, idx_ws,
                                              out + O_ZQ, sums, counts, loss_acc);
    // K4: finalize
    finalize_nt_kernel<<<1, 1024, 0, stream>>>(N_t, counts, loss_acc, out);
    finalize_mt_kernel<<<N_CODES, 256, 0, stream>>>(m_t, sums, counts, out + O_MT);
}
